// BeatDetectionRSNN2_25254407700684
// MI455X (gfx1250) — compile-verified, matrix operand order checked by reading, not run on hardware
//
#include <hip/hip_runtime.h>

// Fused spiking-NN kernel for gfx1250 (MI455X).
// One workgroup per batch element b; 128 threads = 4 wave32.
// Chunked over T in steps of TC=32, with a ping-pong LDS x-tile so global
// loads for chunk c+1 overlap the WMMA GEMM of chunk c:
//   A) issue global_load_b128 for x tile c+1 into registers
//   B) u = x_tile(c) @ W1^T via v_wmma_f32_16x16x32_f16 (W1 frags in registers)
//   C) convert + ds_store tile c+1 into the other LDS buffer
//   D) per-channel integrate-and-fire scan (v1 persistent in registers)
//   E) y = spikes @ W2^T via WMMA on wave 0 (W2 zero-padded to 16 rows)
//   F) thread-0 layer-2 scan (2 accumulators), coalesced s2 store.

typedef __attribute__((ext_vector_type(16))) _Float16 v16h;
typedef __attribute__((ext_vector_type(8)))  _Float16 v8h;
typedef __attribute__((ext_vector_type(4)))  _Float16 v4h;
typedef __attribute__((ext_vector_type(8)))  float    v8f;
typedef __attribute__((ext_vector_type(4)))  float    v4f;

#define T_STEPS 4096
#define D_DIM   128
#define O_DIM   2
#define TC      32      // time-steps per chunk
#define NC      (T_STEPS / TC)
#define XS      136     // lds_x row stride in halfs (16B-aligned rows, bank-spread)
#define US      132     // lds_u row stride in floats (bank-spread for WMMA scatter)
#define SS      136     // lds_s row stride in halfs

// Load one 16-lane-group A/B fragment row: 8 halfs at p and 8 at p+16,
// packed per the CDNA5 16-bit 16x32 VGPR layout (two ds_load_b128).
__device__ __forceinline__ v16h ld_frag_lds(const _Float16* p) {
  v8h lo = *(const v8h*)(p);
  v8h hi = *(const v8h*)(p + 16);
  v16h r;
#pragma unroll
  for (int i = 0; i < 8; ++i) { r[i] = lo[i]; r[8 + i] = hi[i]; }
  return r;
}

__global__ __launch_bounds__(128, 1)
void snn_if2_fused(const float* __restrict__ x, const float* __restrict__ W1,
                   const float* __restrict__ W2, float* __restrict__ out) {
  __shared__ _Float16 lds_x[2][TC * XS];   // ping-pong x tile, f16, [tt][k]
  __shared__ float    lds_u[TC * US];      // GEMM result, f32, [tt][d]
  __shared__ _Float16 lds_s[TC * SS];      // layer-1 spikes, f16, [tt][d]
  __shared__ float    lds_y[TC * O_DIM];   // layer-2 pre-activation
  __shared__ float    lds_s2[TC * O_DIM];  // layer-2 spikes

  const int b    = blockIdx.x;
  const int tid  = threadIdx.x;
  const int lane = tid & 31;
  const int wv   = tid >> 5;
  const int klo  = (lane & 16) ? 8 : 0;    // K offset for this half-wave
  const int nrow = lane & 15;              // M (A) or N (B) row within tile

  // ---- W1 B-fragments, held in registers for the whole kernel.
  // Wave wv owns output channels d in [32*wv, 32*wv+32): 2 N-tiles x 4 K-blocks.
  v16h bw1[2][4];
#pragma unroll
  for (int nt = 0; nt < 2; ++nt) {
    const int d = wv * 32 + nt * 16 + nrow;
    const float* wrow = W1 + d * D_DIM;    // B[k][n] = W1[n][k] -> read row d over k
#pragma unroll
    for (int kb = 0; kb < 4; ++kb) {
      const float* p = wrow + kb * 32 + klo;
      v16h f;
#pragma unroll
      for (int i = 0; i < 8; ++i) { f[i] = (_Float16)p[i]; f[8 + i] = (_Float16)p[16 + i]; }
      bw1[nt][kb] = f;
    }
  }

  // ---- W2 B-fragments (wave 0 only): rows 0,1 real, rows 2..15 zero.
  v16h bw2[4];
  if (wv == 0) {
#pragma unroll
    for (int kb = 0; kb < 4; ++kb) {
      v16h f;
      if (nrow < O_DIM) {
        const float* p = W2 + nrow * D_DIM + kb * 32 + klo;
#pragma unroll
        for (int i = 0; i < 8; ++i) { f[i] = (_Float16)p[i]; f[8 + i] = (_Float16)p[16 + i]; }
      } else {
#pragma unroll
        for (int i = 0; i < 16; ++i) f[i] = (_Float16)0.0f;
      }
      bw2[kb] = f;
    }
  }

  float v1 = 0.0f;                 // layer-1 membrane, channel d = tid
  float v20 = 0.0f, v21 = 0.0f;    // layer-2 membranes (thread 0)
  const float* xb  = x + (size_t)b * T_STEPS * D_DIM;
  float*      outb = out + (size_t)b * T_STEPS * O_DIM;

  // ---- prologue: stage tile 0 into buffer 0.
  {
    const float* xt = xb;
#pragma unroll
    for (int i = 0; i < 8; ++i) {
      const int j  = i * 128 + tid;        // float4 index
      v4f v = ((const v4f*)xt)[j];
      const int tt = j >> 5;
      const int kk = (j & 31) * 4;
      *(v4h*)&lds_x[0][tt * XS + kk] =
          (v4h){(_Float16)v[0], (_Float16)v[1], (_Float16)v[2], (_Float16)v[3]};
    }
    __builtin_prefetch(xt + TC * D_DIM + tid * 32, 0, 0);
  }
  __syncthreads();

  for (int c = 0; c < NC; ++c) {
    const _Float16* bx_cur = lds_x[c & 1];

    // ---- A) issue global loads for tile c+1 (consumed after the GEMM).
    v4f xr[8];
    if (c + 1 < NC) {
      const float* xn = xb + (c + 1) * TC * D_DIM;
#pragma unroll
      for (int i = 0; i < 8; ++i)
        xr[i] = ((const v4f*)xn)[i * 128 + tid];
      if (c + 2 < NC)
        __builtin_prefetch(xn + TC * D_DIM + tid * 32, 0, 0);  // 2 chunks ahead
    }

    // ---- B) GEMM: u[tt][d] = sum_k x[tt][k] * W1[d][k], 16 WMMAs per wave.
#pragma unroll
    for (int mt = 0; mt < 2; ++mt) {
      v16h a[4];
      const int arow = mt * 16 + nrow;
#pragma unroll
      for (int kb = 0; kb < 4; ++kb)
        a[kb] = ld_frag_lds(&bx_cur[arow * XS + kb * 32 + klo]);
#pragma unroll
      for (int nt = 0; nt < 2; ++nt) {
        v8f acc = {0.f, 0.f, 0.f, 0.f, 0.f, 0.f, 0.f, 0.f};
#pragma unroll
        for (int kb = 0; kb < 4; ++kb)
          acc = __builtin_amdgcn_wmma_f32_16x16x32_f16(
              false, a[kb], false, bw1[nt][kb], (short)0, acc, false, false);
        const int d   = wv * 32 + nt * 16 + nrow;
        const int ttb = mt * 16 + ((lane & 16) ? 8 : 0);
#pragma unroll
        for (int r = 0; r < 8; ++r)
          lds_u[(ttb + r) * US + d] = acc[r];
      }
    }

    // ---- C) convert + store tile c+1 into the other buffer.
    if (c + 1 < NC) {
      _Float16* bx_nxt = lds_x[(c + 1) & 1];
#pragma unroll
      for (int i = 0; i < 8; ++i) {
        const int j  = i * 128 + tid;
        const int tt = j >> 5;
        const int kk = (j & 31) * 4;
        *(v4h*)&bx_nxt[tt * XS + kk] =
            (v4h){(_Float16)xr[i][0], (_Float16)xr[i][1],
                  (_Float16)xr[i][2], (_Float16)xr[i][3]};
      }
    }
    __syncthreads();   // u visible for scan; tile c+1 visible for next GEMM

    // ---- D) layer-1 IF scan: thread tid owns channel d = tid.
#pragma unroll 4
    for (int tt = 0; tt < TC; ++tt) {
      float v  = v1 + lds_u[tt * US + tid];
      bool  sp = (v >= 1.0f);
      lds_s[tt * SS + tid] = sp ? (_Float16)1.0f : (_Float16)0.0f;
      v1 = sp ? 0.0f : v;                 // hard reset to 0
    }
    __syncthreads();

    // ---- E) y = S @ W2^T via WMMA on wave 0 (8 WMMAs).
    if (wv == 0) {
#pragma unroll
      for (int mt = 0; mt < 2; ++mt) {
        v8f acc = {0.f, 0.f, 0.f, 0.f, 0.f, 0.f, 0.f, 0.f};
        const int arow = mt * 16 + nrow;
#pragma unroll
        for (int kb = 0; kb < 4; ++kb) {
          v16h a = ld_frag_lds(&lds_s[arow * SS + kb * 32 + klo]);
          acc = __builtin_amdgcn_wmma_f32_16x16x32_f16(
              false, a, false, bw2[kb], (short)0, acc, false, false);
        }
        if (nrow < O_DIM) {               // only columns N=0,1 are real
          const int ttb = mt * 16 + ((lane & 16) ? 8 : 0);
#pragma unroll
          for (int r = 0; r < 8; ++r)
            lds_y[(ttb + r) * O_DIM + nrow] = acc[r];
        }
      }
    }
    __syncthreads();

    // ---- F) layer-2 IF scan (2 accumulators, thread 0).
    if (tid == 0) {
#pragma unroll 8
      for (int tt = 0; tt < TC; ++tt) {
        float a0 = v20 + lds_y[tt * 2 + 0];
        float a1 = v21 + lds_y[tt * 2 + 1];
        bool  s0 = (a0 >= 1.0f), s1 = (a1 >= 1.0f);
        lds_s2[tt * 2 + 0] = s0 ? 1.0f : 0.0f;
        lds_s2[tt * 2 + 1] = s1 ? 1.0f : 0.0f;
        v20 = s0 ? 0.0f : a0;
        v21 = s1 ? 0.0f : a1;
      }
    }
    __syncthreads();

    // ---- coalesced store of 64 output floats (32 t x 2 o).
    if (tid < TC * O_DIM)
      outb[c * TC * O_DIM + tid] = lds_s2[tid];
  }
}

extern "C" void kernel_launch(void* const* d_in, const int* in_sizes, int n_in,
                              void* d_out, int out_size, void* d_ws, size_t ws_size,
                              hipStream_t stream) {
  (void)n_in; (void)out_size; (void)d_ws; (void)ws_size;
  const float* x  = (const float*)d_in[0];   // [B, T, D] f32
  const float* W1 = (const float*)d_in[1];   // [D, D]    f32
  const float* W2 = (const float*)d_in[2];   // [O, D]    f32
  float* out = (float*)d_out;                // [B, T, O] f32
  const int B = in_sizes[0] / (T_STEPS * D_DIM);
  snn_if2_fused<<<B, 128, 0, stream>>>(x, W1, W2, out);
}
